// SpectralCrossAttentionBlock_58548994179741
// MI455X (gfx1250) — matrix operand order repaired and checked by reading, not compiled, not run on hardware
//
#include <hip/hip_runtime.h>
#include <math.h>

// ---------------- gfx1250 feature detection ----------------
#if defined(__has_builtin)
#  if __has_builtin(__builtin_amdgcn_global_load_async_to_lds_b32)
#    define SCA_HAS_ASYNC 1
#  endif
#endif
#ifndef SCA_HAS_ASYNC
#  define SCA_HAS_ASYNC 0
#endif

typedef __attribute__((ext_vector_type(16))) _Float16 v16h;
typedef __attribute__((ext_vector_type(8)))  _Float16 v8h;
typedef __attribute__((ext_vector_type(8)))  float    v8f;

#define NSEG 50

struct SCAState {
  unsigned fmin_k, fmax_k, sfmin_k, sfmax_k, magmax_k;
  unsigned pfx[2], rnk[2];
  float    kval[2];
  float    thresh;
  unsigned hist0[2048];
  unsigned h1[2][2048];
  unsigned h2[2][1024];
  float seg_sum[NSEG * 3];
  float seg_cnt[NSEG];
  float seg_mean[NSEG * 3];
  float seg_valid[NSEG];
};

// monotonic float <-> uint key (total order incl. negatives)
__device__ __forceinline__ unsigned fkey(float f) {
  unsigned u = __float_as_uint(f);
  return (u & 0x80000000u) ? ~u : (u | 0x80000000u);
}
__device__ __forceinline__ float funkey(unsigned k) {
  unsigned u = (k & 0x80000000u) ? (k & 0x7fffffffu) : ~k;
  return __uint_as_float(u);
}
__device__ __forceinline__ float clamp01(float v) { return fminf(fmaxf(v, 0.f), 1.f); }
__device__ __forceinline__ int   iclamp(int v, int lo, int hi) { return v < lo ? lo : (v > hi ? hi : v); }

// async global->LDS staging (gfx1250 path), plain-load fallback
__device__ __forceinline__ void async_copy_f32(const float* g, float* l) {
#if SCA_HAS_ASYNC
  __attribute__((address_space(3))) float* l3 =
      (__attribute__((address_space(3))) float*)l;            // addrspacecast
  __builtin_amdgcn_global_load_async_to_lds_b32(
      (__attribute__((address_space(1))) int*)(unsigned long long)g,
      (__attribute__((address_space(3))) int*)l3, 0, 0);
#else
  *l = *g;
#endif
}
__device__ __forceinline__ void async_wait_all() {
#if SCA_HAS_ASYNC
#  if __has_builtin(__builtin_amdgcn_s_wait_asynccnt)
  __builtin_amdgcn_s_wait_asynccnt(0);
#  else
  asm volatile("s_wait_asynccnt 0" ::: "memory");
#  endif
#endif
}

// ---------------- init ----------------
__global__ void sca_init(SCAState* st, unsigned k0, unsigned k1) {
  st->fmin_k = 0xFFFFFFFFu; st->fmax_k = 0u;
  st->sfmin_k = 0xFFFFFFFFu; st->sfmax_k = 0u;
  st->magmax_k = 0u;
  st->pfx[0] = st->pfx[1] = 0u;
  st->rnk[0] = k0; st->rnk[1] = k1;
}

// ---------------- global min/max of fiedler & source_fiedler ----------------
__global__ __launch_bounds__(256) void sca_minmax(const float* __restrict__ f,
                                                  const float* __restrict__ sf,
                                                  int n, SCAState* st) {
  unsigned fmn = 0xFFFFFFFFu, fmx = 0u, smn = 0xFFFFFFFFu, smx = 0u;
  int stride = gridDim.x * blockDim.x;
  for (int i = blockIdx.x * blockDim.x + threadIdx.x; i < n; i += stride) {
    __builtin_prefetch(f + i + stride, 0, 0);
    __builtin_prefetch(sf + i + stride, 0, 0);
    unsigned kf = fkey(f[i]);  fmn = min(fmn, kf); fmx = max(fmx, kf);
    unsigned ks = fkey(sf[i]); smn = min(smn, ks); smx = max(smx, ks);
  }
  __shared__ unsigned ra[256], rb[256], rc[256], rd[256];
  int t = threadIdx.x;
  ra[t] = fmn; rb[t] = fmx; rc[t] = smn; rd[t] = smx;
  __syncthreads();
  for (int s = 128; s > 0; s >>= 1) {
    if (t < s) {
      ra[t] = min(ra[t], ra[t + s]); rb[t] = max(rb[t], rb[t + s]);
      rc[t] = min(rc[t], rc[t + s]); rd[t] = max(rd[t], rd[t + s]);
    }
    __syncthreads();
  }
  if (t == 0) {
    atomicMin(&st->fmin_k, ra[0]);  atomicMax(&st->fmax_k, rb[0]);
    atomicMin(&st->sfmin_k, rc[0]); atomicMax(&st->sfmax_k, rd[0]);
  }
}

// ---------------- luminance -> Sobel magnitude (+ global max) ----------------
__global__ __launch_bounds__(256) void sca_mag(const float* __restrict__ img,
                                               float* __restrict__ mag,
                                               SCAState* st, int H, int W) {
  __shared__ float g[34][36];
  int x0 = blockIdx.x * 32, y0 = blockIdx.y * 32;
  for (int idx = threadIdx.x; idx < 34 * 34; idx += 256) {
    int r = idx / 34, c = idx % 34;
    int gy = y0 - 1 + r, gx = x0 - 1 + c;
    float v = 0.f;
    if (gy >= 0 && gy < H && gx >= 0 && gx < W) {
      const float* p = img + 3 * ((size_t)gy * W + gx);
      v = 0.299f * p[0] + 0.587f * p[1] + 0.114f * p[2];
    }
    g[r][c] = v;
  }
  __syncthreads();
  float lmax = 0.f;
  for (int p = 0; p < 4; p++) {
    int li = threadIdx.x + 256 * p;
    int ty = li >> 5, tx = li & 31;
    float gx_ = (-g[ty][tx] + g[ty][tx + 2] - 2.f * g[ty + 1][tx] + 2.f * g[ty + 1][tx + 2]
                 - g[ty + 2][tx] + g[ty + 2][tx + 2]) * 0.125f;
    float gy_ = (-g[ty][tx] - 2.f * g[ty][tx + 1] - g[ty][tx + 2]
                 + g[ty + 2][tx] + 2.f * g[ty + 2][tx + 1] + g[ty + 2][tx + 2]) * 0.125f;
    float m = sqrtf(gx_ * gx_ + gy_ * gy_ + 1e-12f);
    int yy = y0 + ty, xx = x0 + tx;
    if (yy < H && xx < W) mag[(size_t)yy * W + xx] = m;
    lmax = fmaxf(lmax, m);
  }
  __shared__ float red[256];
  red[threadIdx.x] = lmax;
  __syncthreads();
  for (int s = 128; s > 0; s >>= 1) {
    if (threadIdx.x < s) red[threadIdx.x] = fmaxf(red[threadIdx.x], red[threadIdx.x + s]);
    __syncthreads();
  }
  if (threadIdx.x == 0) atomicMax(&st->magmax_k, fkey(red[0]));
}

// ---------------- radix-select histogram passes (exact 40th percentile) ------
__global__ __launch_bounds__(256) void sca_hist(const float* __restrict__ f, int n,
                                                SCAState* st, int pass) {
  __shared__ unsigned sh[4096];
  for (int i = threadIdx.x; i < 4096; i += 256) sh[i] = 0u;
  __syncthreads();
  unsigned p0 = st->pfx[0], p1 = st->pfx[1];
  int stride = gridDim.x * blockDim.x;
  for (int i = blockIdx.x * blockDim.x + threadIdx.x; i < n; i += stride) {
    __builtin_prefetch(f + i + stride, 0, 0);
    unsigned k = fkey(f[i]);
    if (pass == 0) {
      atomicAdd(&sh[k >> 21], 1u);
    } else if (pass == 1) {
      if ((k >> 21) == p0) atomicAdd(&sh[(k >> 10) & 2047u], 1u);
      if ((k >> 21) == p1) atomicAdd(&sh[2048u + ((k >> 10) & 2047u)], 1u);
    } else {
      if ((k >> 10) == p0) atomicAdd(&sh[k & 1023u], 1u);
      if ((k >> 10) == p1) atomicAdd(&sh[2048u + (k & 1023u)], 1u);
    }
  }
  __syncthreads();
  if (pass == 0) {
    for (int i = threadIdx.x; i < 2048; i += 256)
      if (sh[i]) atomicAdd(&st->hist0[i], sh[i]);
  } else if (pass == 1) {
    for (int i = threadIdx.x; i < 2048; i += 256) {
      if (sh[i])        atomicAdd(&st->h1[0][i], sh[i]);
      if (sh[2048 + i]) atomicAdd(&st->h1[1][i], sh[2048 + i]);
    }
  } else {
    for (int i = threadIdx.x; i < 1024; i += 256) {
      if (sh[i])        atomicAdd(&st->h2[0][i], sh[i]);
      if (sh[2048 + i]) atomicAdd(&st->h2[1][i], sh[2048 + i]);
    }
  }
}

__global__ void sca_scan(SCAState* st, int pass, float frac) {
  for (int r = 0; r < 2; r++) {
    unsigned rem = st->rnk[r];
    unsigned cum = 0;
    if (pass == 0) {
      for (int b = 0; b < 2048; b++) {
        unsigned c = st->hist0[b];
        if (rem < cum + c) { st->pfx[r] = (unsigned)b; st->rnk[r] = rem - cum; break; }
        cum += c;
      }
    } else if (pass == 1) {
      for (int b = 0; b < 2048; b++) {
        unsigned c = st->h1[r][b];
        if (rem < cum + c) { st->pfx[r] = (st->pfx[r] << 11) | (unsigned)b; st->rnk[r] = rem - cum; break; }
        cum += c;
      }
    } else {
      for (int b = 0; b < 1024; b++) {
        unsigned c = st->h2[r][b];
        if (rem < cum + c) { st->kval[r] = funkey((st->pfx[r] << 10) | (unsigned)b); break; }
        cum += c;
      }
    }
  }
  if (pass == 2) st->thresh = st->kval[0] + frac * (st->kval[1] - st->kval[0]);
}

// ---------------- segment sums of source over source_fiedler bins ------------
__global__ __launch_bounds__(256) void sca_seg(const float* __restrict__ src,
                                               const float* __restrict__ sf,
                                               int n, SCAState* st) {
  __shared__ float ssum[NSEG * 3];
  __shared__ float scnt[NSEG];
  for (int i = threadIdx.x; i < NSEG * 3; i += 256) ssum[i] = 0.f;
  for (int i = threadIdx.x; i < NSEG; i += 256) scnt[i] = 0.f;
  __syncthreads();
  float smin = funkey(st->sfmin_k), smax = funkey(st->sfmax_k);
  float scale = (float)NSEG / (smax - smin + 1e-8f);
  int stride = gridDim.x * blockDim.x;
  for (int i = blockIdx.x * blockDim.x + threadIdx.x; i < n; i += stride) {
    __builtin_prefetch(sf + i + stride, 0, 0);
    float v = sf[i];
    int b = iclamp((int)((v - smin) * scale), 0, NSEG - 1);
    const float* p = src + 3 * (size_t)i;
    atomicAdd(&ssum[b * 3 + 0], p[0]);
    atomicAdd(&ssum[b * 3 + 1], p[1]);
    atomicAdd(&ssum[b * 3 + 2], p[2]);
    atomicAdd(&scnt[b], 1.f);
  }
  __syncthreads();
  for (int i = threadIdx.x; i < NSEG * 3; i += 256)
    if (ssum[i] != 0.f) atomicAdd(&st->seg_sum[i], ssum[i]);
  for (int i = threadIdx.x; i < NSEG; i += 256)
    if (scnt[i] != 0.f) atomicAdd(&st->seg_cnt[i], scnt[i]);
}

__global__ void sca_segfin(SCAState* st) {
  int b = threadIdx.x;
  if (b < NSEG) {
    float c = st->seg_cnt[b];
    float d = fmaxf(c, 1.f);
    st->seg_mean[b * 3 + 0] = st->seg_sum[b * 3 + 0] / d;
    st->seg_mean[b * 3 + 1] = st->seg_sum[b * 3 + 1] / d;
    st->seg_mean[b * 3 + 2] = st->seg_sum[b * 3 + 2] / d;
    st->seg_valid[b] = (c >= 20.f) ? 1.f : 0.f;
  }
}

// ---------------- 5x5 gated Gaussian via WMMA (banded GEMM) ------------------
// Horizontal pass: Out(16x16) = A(16x32 f16 tile rows) x B(32x16 banded 1D
// Gaussian), accumulated in f32. Vertical 5-tap done in VALU from LDS.
// cm has 48 columns (zero-padded) so A fragments are unconditional aligned
// v8h LDS loads; hs has 48 rows so D stores are unconditional.
__global__ __launch_bounds__(256) void sca_dil(const float* __restrict__ mag,
                                               const float* __restrict__ fied,
                                               float* __restrict__ dil,
                                               const SCAState* st, int H, int W) {
  __shared__ __align__(16) _Float16 cm[36][48];  // gated contours, halo 2 + pad
  __shared__ float hs[48][34];                   // horizontal-filtered rows
  int x0 = blockIdx.x * 32, y0 = blockIdx.y * 32;
  float maxmag = funkey(st->magmax_k);
  float invm = 1.f / (maxmag + 1e-8f);
  float th = st->thresh;  // sigmoid(10*(f-th)) > 0.5  <=>  f > th

  for (int idx = threadIdx.x; idx < 36 * 48; idx += 256) {
    int r = idx / 48, c = idx % 48;
    int gy = y0 - 2 + r, gx = x0 - 2 + c;
    float v = 0.f;  // SAME conv zero-pad; pad cols stay finite-zero
    if (c < 36 && gy >= 0 && gy < H && gx >= 0 && gx < W) {
      size_t o = (size_t)gy * W + gx;
      v = (fied[o] > th) ? mag[o] * invm : 0.f;
    }
    cm[r][c] = (_Float16)v;
  }
  __syncthreads();

  // 1D Gaussian, sigma = 0.6 * 2 = 1.2 (product of normalized 1Ds == 2D kernel)
  float e0 = __expf(-4.f / 2.88f), e1 = __expf(-1.f / 2.88f);
  float isum = 1.f / (2.f * e0 + 2.f * e1 + 1.f);
  float ge = e0 * isum, gm = e1 * isum, gc = isum;

  int wave = threadIdx.x >> 5, lid = threadIdx.x & 31;
  if (wave < 6) {                       // 6 tiles: 3 row-strips x 2 col-strips
    int tr = wave >> 1, tc = wave & 1;
    int R0 = 16 * tr, C0 = 16 * tc;
    int M = lid & 15;
    int ar = min(R0 + M, 35);           // rows >=36 computed-then-discarded
    int kf = (lid < 16) ? 0 : 8;        // f16 A layout (16x32)
    // halves 0..7 : K = kf..kf+7 ; halves 8..15 : K = kf+16..kf+23
    const v8h* rowp = (const v8h*)&cm[ar][0];
    v8h alo = rowp[(C0 + kf) >> 3];
    v8h ahi = rowp[(C0 + kf + 16) >> 3];
    v16h a = __builtin_shufflevector(alo, ahi, 0, 1, 2, 3, 4, 5, 6, 7,
                                     8, 9, 10, 11, 12, 13, 14, 15);
    int n = lid & 15;
    int kb = (lid < 16) ? 0 : 8;        // f16 B layout (32x16)
    v16h b;
    for (int h = 0; h < 16; h++) {
      int dk = (kb + h + (h >= 8 ? 8 : 0)) - n;  // B[k][n] = g[k-n], band 0..4
      float wv = 0.f;
      if (dk == 0 || dk == 4) wv = ge;
      else if (dk == 1 || dk == 3) wv = gm;
      else if (dk == 2) wv = gc;
      b[h] = (_Float16)wv;
    }
    v8f cacc = {};
    cacc = __builtin_amdgcn_wmma_f32_16x16x32_f16(
        false, a, false, b, (short)0, cacc, false, false);
    int rbase = R0 + ((lid < 16) ? 0 : 8);
    for (int v = 0; v < 8; v++)         // D layout: row = v (+8 for hi lanes)
      hs[rbase + v][C0 + n] = cacc[v];
  }
  __syncthreads();

  for (int p = 0; p < 4; p++) {
    int li = threadIdx.x + 256 * p;
    int ty = li >> 5, tx = li & 31;
    float d = ge * hs[ty][tx] + gm * hs[ty + 1][tx] + gc * hs[ty + 2][tx]
            + gm * hs[ty + 3][tx] + ge * hs[ty + 4][tx];
    int yy = y0 + ty, xx = x0 + tx;
    if (yy < H && xx < W) dil[(size_t)yy * W + xx] = d;
  }
}

// ---------------- helpers for final composite --------------------------------
__device__ __forceinline__ void bilinear_img(const float* img, int H, int W,
                                             float yy, float xx, float* o3) {
  float y0f = fminf(fmaxf(floorf(yy), 0.f), (float)(H - 1));
  float x0f = fminf(fmaxf(floorf(xx), 0.f), (float)(W - 1));
  int y0 = (int)y0f, x0 = (int)x0f;
  int y1 = min(y0 + 1, H - 1), x1 = min(x0 + 1, W - 1);
  float wy = fminf(fmaxf(yy - y0f, 0.f), 1.f);
  float wx = fminf(fmaxf(xx - x0f, 0.f), 1.f);
  const float* p00 = img + 3 * ((size_t)y0 * W + x0);
  const float* p01 = img + 3 * ((size_t)y0 * W + x1);
  const float* p10 = img + 3 * ((size_t)y1 * W + x0);
  const float* p11 = img + 3 * ((size_t)y1 * W + x1);
  for (int c = 0; c < 3; c++)
    o3[c] = (1.f - wy) * ((1.f - wx) * p00[c] + wx * p01[c])
          + wy * ((1.f - wx) * p10[c] + wx * p11[c]);
}

__device__ __forceinline__ void seg_lookup(const float* fied, const SCAState* st,
                                           int W, int yi, int xi,
                                           float fmin, float fscale, float* c3) {
  float fv = fied[(size_t)yi * W + xi];
  int b = iclamp((int)((fv - fmin) * fscale), 0, NSEG - 1);
  c3[0] = st->seg_mean[b * 3 + 0];
  c3[1] = st->seg_mean[b * 3 + 1];
  c3[2] = st->seg_mean[b * 3 + 2];
}

// ---------------- final fused composite --------------------------------------
__global__ __launch_bounds__(256) void sca_final(const float* __restrict__ img,
                                                 const float* __restrict__ fied,
                                                 const float* __restrict__ mag,
                                                 const float* __restrict__ dil,
                                                 const SCAState* st,
                                                 float* __restrict__ out,
                                                 int H, int W) {
  __shared__ float ft[34][36];  // fiedler, halo 1 (zero-padded Sobel)
  __shared__ float ms[32][33];  // mag, center tile (async-staged)
  int x0 = blockIdx.x * 32, y0 = blockIdx.y * 32;

  for (int idx = threadIdx.x; idx < 34 * 34; idx += 256) {
    int r = idx / 34, c = idx % 34;
    int gy = y0 - 1 + r, gx = x0 - 1 + c;
    ft[r][c] = (gy >= 0 && gy < H && gx >= 0 && gx < W)
                   ? fied[(size_t)gy * W + gx] : 0.f;
  }
  for (int p = 0; p < 4; p++) {
    int li = threadIdx.x + 256 * p;
    int ty = li >> 5, tx = li & 31;
    int gy = y0 + ty, gx = x0 + tx;
    if (gy < H && gx < W) async_copy_f32(&mag[(size_t)gy * W + gx], &ms[ty][tx]);
    else                  ms[ty][tx] = 0.f;
  }
  async_wait_all();
  __syncthreads();

  float th = st->thresh;
  float invm = 1.f / (funkey(st->magmax_k) + 1e-8f);
  float fmin = funkey(st->fmin_k), fmax = funkey(st->fmax_k);
  float fscale = (float)NSEG / (fmax - fmin + 1e-8f);
  float cy = (H - 1) * 0.5f, cx = (W - 1) * 0.5f;
  float cr = cosf(0.03f), sr = sinf(0.03f);

  for (int p = 0; p < 4; p++) {
    int li = threadIdx.x + 256 * p;
    int ty = li >> 5, tx = li & 31;
    int y = y0 + ty, x = x0 + tx;
    if (y >= H || x >= W) continue;

    float f = ft[ty + 1][tx + 1];
    float gate = 1.f / (1.f + __expf(-10.f * (f - th)));
    float cont = ms[ty][tx] * invm;
    float d = dil[(size_t)y * W + x];
    float alpha = (d > 0.1f) ? clamp01(d) : 0.f;   // fill * clip(dil)

    // rotated resample (0.03 rad about image center, clamped bilinear)
    float yv = (float)y - cy, xv = (float)x - cx;
    float r3[3];
    bilinear_img(img, H, W, cr * yv - sr * xv + cy, sr * yv + cr * xv + cx, r3);
    float modul = 1.f + 0.3f * tanhf(f);
    const float* ip = img + 3 * ((size_t)y * W + x);
    float o0 = clamp01(ip[0] + alpha * (r3[0] * modul - ip[0]));
    float o1 = clamp01(ip[1] + alpha * (r3[1] * modul - ip[1]));
    float o2 = clamp01(ip[2] + alpha * (r3[2] * modul - ip[2]));

    // Sobel of fiedler -> warped segment-transfer sample
    float gxf = (-ft[ty][tx] + ft[ty][tx + 2] - 2.f * ft[ty + 1][tx] + 2.f * ft[ty + 1][tx + 2]
                 - ft[ty + 2][tx] + ft[ty + 2][tx + 2]) * 0.125f;
    float gyf = (-ft[ty][tx] - 2.f * ft[ty][tx + 1] - ft[ty][tx + 2]
                 + ft[ty + 2][tx] + 2.f * ft[ty + 2][tx + 1] + ft[ty + 2][tx + 2]) * 0.125f;
    float sy = (float)y + 7.f + 20.f * tanhf(gyf);
    float sx = (float)x + 20.f * tanhf(gxf);

    float y0f = fminf(fmaxf(floorf(sy), 0.f), (float)(H - 1));
    float x0f = fminf(fmaxf(floorf(sx), 0.f), (float)(W - 1));
    int yi0 = (int)y0f, xi0 = (int)x0f;
    int yi1 = min(yi0 + 1, H - 1), xi1 = min(xi0 + 1, W - 1);
    float wy = fminf(fmaxf(sy - y0f, 0.f), 1.f);
    float wx = fminf(fmaxf(sx - x0f, 0.f), 1.f);
    float c00[3], c01[3], c10[3], c11[3];
    seg_lookup(fied, st, W, yi0, xi0, fmin, fscale, c00);
    seg_lookup(fied, st, W, yi0, xi1, fmin, fscale, c01);
    seg_lookup(fied, st, W, yi1, xi0, fmin, fscale, c10);
    seg_lookup(fied, st, W, yi1, xi1, fmin, fscale, c11);

    int bc = iclamp((int)((f - fmin) * fscale), 0, NSEG - 1);
    float vmask = st->seg_valid[bc];
    float w = clamp01((1.f - gate) * vmask * (1.f - cont));

    float* op = out + 3 * ((size_t)y * W + x);
    for (int c = 0; c < 3; c++) {
      float sh = (1.f - wy) * ((1.f - wx) * c00[c] + wx * c01[c])
               + wy * ((1.f - wx) * c10[c] + wx * c11[c]);
      float oc = (c == 0) ? o0 : (c == 1) ? o1 : o2;
      op[c] = clamp01(oc * (1.f - w) + sh * w);
    }
  }
}

// ---------------- launcher ----------------
extern "C" void kernel_launch(void* const* d_in, const int* in_sizes, int n_in,
                              void* d_out, int out_size, void* d_ws, size_t ws_size,
                              hipStream_t stream) {
  if (n_in < 4) return;
  const float* img   = (const float*)d_in[0];
  const float* fied  = (const float*)d_in[1];
  const float* src   = (const float*)d_in[2];
  const float* sfied = (const float*)d_in[3];
  float* out = (float*)d_out;

  int N = in_sizes[1];
  int H = (int)(sqrt((double)N) + 0.5);
  while ((long long)H * H > (long long)N) --H;
  while ((long long)(H + 1) * (H + 1) <= (long long)N) ++H;
  int W = N / H;

  float* mag = (float*)d_ws;
  float* dil = mag + N;
  SCAState* st = (SCAState*)(dil + N);
  (void)hipMemsetAsync(st, 0, sizeof(SCAState), stream);

  double q = 0.40 * (double)(N - 1);          // PCT=40, linear interpolation
  unsigned k0 = (unsigned)q;
  float frac = (float)(q - (double)k0);
  unsigned k1 = (k0 + 1 < (unsigned)N) ? k0 + 1 : k0;

  sca_init<<<1, 1, 0, stream>>>(st, k0, k1);

  const int RB = 240;
  sca_minmax<<<RB, 256, 0, stream>>>(fied, sfied, N, st);

  dim3 g2((W + 31) / 32, (H + 31) / 32);
  sca_mag<<<g2, 256, 0, stream>>>(img, mag, st, H, W);

  sca_hist<<<RB, 256, 0, stream>>>(fied, N, st, 0);
  sca_scan<<<1, 1, 0, stream>>>(st, 0, frac);
  sca_hist<<<RB, 256, 0, stream>>>(fied, N, st, 1);
  sca_scan<<<1, 1, 0, stream>>>(st, 1, frac);
  sca_hist<<<RB, 256, 0, stream>>>(fied, N, st, 2);
  sca_scan<<<1, 1, 0, stream>>>(st, 2, frac);

  sca_seg<<<RB, 256, 0, stream>>>(src, sfied, N, st);
  sca_segfin<<<1, 64, 0, stream>>>(st);

  sca_dil<<<g2, 256, 0, stream>>>(mag, fied, dil, st, H, W);
  sca_final<<<g2, 256, 0, stream>>>(img, fied, mag, dil, st, out, H, W);
}
